// predictNet_85555748536778
// MI455X (gfx1250) — matrix operand order repaired and checked by reading, not compiled, not run on hardware
//
#include <hip/hip_runtime.h>

typedef __attribute__((ext_vector_type(16))) _Float16 v16h;
typedef __attribute__((ext_vector_type(8)))  float    v8f;
typedef __attribute__((ext_vector_type(4)))  int      vi4;
typedef __attribute__((address_space(1))) vi4 gvi4;   // global int4
typedef __attribute__((address_space(3))) vi4 svi4;   // LDS int4

#define L_TOTAL 262144
#define L_MASK  (L_TOTAL - 1)
#define LN_EPS  1e-5f
#define AVG_K   151
#define AVG_HALF 75

#if defined(__has_builtin)
#if __has_builtin(__builtin_amdgcn_global_load_async_to_lds_b128)
#define HAVE_ASYNC_LDS 1
#endif
#endif

// ---------------------------------------------------------------------------
// Stats: partial sums of (v, v^2) then finalize to (mean, rstd)
// ---------------------------------------------------------------------------
__global__ __launch_bounds__(256) void stat_partial_kernel(
    const float* __restrict__ src, int n, float* __restrict__ part) {
  __shared__ float s1[256], s2[256];
  float a = 0.f, b = 0.f;
  for (int i = blockIdx.x * 256 + threadIdx.x; i < n; i += gridDim.x * 256) {
    float v = src[i];
    a += v;
    b += v * v;
  }
  s1[threadIdx.x] = a; s2[threadIdx.x] = b;
  __syncthreads();
  for (int s = 128; s > 0; s >>= 1) {
    if (threadIdx.x < s) {
      s1[threadIdx.x] += s1[threadIdx.x + s];
      s2[threadIdx.x] += s2[threadIdx.x + s];
    }
    __syncthreads();
  }
  if (threadIdx.x == 0) {
    part[blockIdx.x * 2 + 0] = s1[0];
    part[blockIdx.x * 2 + 1] = s2[0];
  }
}

__global__ __launch_bounds__(256) void stat_final_kernel(
    const float* __restrict__ part, int nblocks, float invN,
    float* __restrict__ stats) {
  __shared__ float s1[256], s2[256];
  float a = 0.f, b = 0.f;
  for (int i = threadIdx.x; i < nblocks; i += 256) {
    a += part[i * 2 + 0];
    b += part[i * 2 + 1];
  }
  s1[threadIdx.x] = a; s2[threadIdx.x] = b;
  __syncthreads();
  for (int s = 128; s > 0; s >>= 1) {
    if (threadIdx.x < s) {
      s1[threadIdx.x] += s1[threadIdx.x + s];
      s2[threadIdx.x] += s2[threadIdx.x + s];
    }
    __syncthreads();
  }
  if (threadIdx.x == 0) {
    float m   = s1[0] * invN;
    float var = s2[0] * invN - m * m;
    stats[0] = m;
    stats[1] = rsqrtf(var + LN_EPS);
  }
}

// ---------------------------------------------------------------------------
// v = (x - mean) * rstd  -> f32 and f16 copies
// ---------------------------------------------------------------------------
__global__ __launch_bounds__(256) void normalize_kernel(
    const float* __restrict__ x, const float* __restrict__ stats,
    float* __restrict__ vf, _Float16* __restrict__ vh) {
  int i = blockIdx.x * 256 + threadIdx.x;
  float v = (x[i] - stats[0]) * stats[1];
  vf[i] = v;
  vh[i] = (_Float16)v;
}

// ---------------------------------------------------------------------------
// AvgPool1d(kernel=151, stride=1, pad=75, count_include_pad) on v and v^2
// ---------------------------------------------------------------------------
__global__ __launch_bounds__(256) void avgpool_kernel(
    const float* __restrict__ vf, float* __restrict__ avgf,
    _Float16* __restrict__ avgh, float* __restrict__ energy) {
  __shared__ float sv[256 + 2 * AVG_HALF];
  int base = blockIdx.x * 256;
  for (int i = threadIdx.x; i < 256 + 2 * AVG_HALF; i += 256) {
    int g = base - AVG_HALF + i;
    sv[i] = (g >= 0 && g < L_TOTAL) ? vf[g] : 0.f;
  }
  __syncthreads();
  float s1 = 0.f, s2 = 0.f;
  for (int t = 0; t < AVG_K; ++t) {
    float v = sv[threadIdx.x + t];
    s1 += v;
    s2 += v * v;
  }
  int j = base + threadIdx.x;
  float a = s1 * (1.f / AVG_K);
  avgf[j]   = a;
  avgh[j]   = (_Float16)a;
  energy[j] = s2 * (1.f / AVG_K);
}

// ---------------------------------------------------------------------------
// Pack Wa [H x D] into per-fragment WMMA A layout (16x32 f16 tiles).
// ---------------------------------------------------------------------------
template <int D, int H, int KT, int MT>
__global__ __launch_bounds__(256) void pack_a_kernel(
    const float* __restrict__ Wa, _Float16* __restrict__ apack) {
  int idx = blockIdx.x * 256 + threadIdx.x;
  if (idx >= MT * KT * 512) return;
  int t    = idx & 15;
  int lane = (idx >> 4) & 31;
  int frag = idx >> 9;
  int kt = frag % KT, mt = frag / KT;
  int r = t >> 1, h = t & 1;
  int bl = (lane >= 16) ? 8 : 0;
  int k  = (r < 4) ? (bl + 2 * r + h) : (16 + bl + 2 * (r - 4) + h);
  int f  = kt * 32 + k;
  int m  = mt * 16 + (lane & 15);
  float val = (m < H && f < D) ? Wa[m * D + f] : 0.f;
  apack[idx] = (_Float16)val;
}

// ---------------------------------------------------------------------------
// wEff = mean over p rows of Wb (zero-padded to HP); bap = padded ba;
// beff = mean(bb)
// ---------------------------------------------------------------------------
template <int H, int HP, int P>
__global__ __launch_bounds__(256) void weff_kernel(
    const float* __restrict__ Wb, const float* __restrict__ bb,
    const float* __restrict__ ba, float* __restrict__ weff,
    float* __restrict__ bap, float* __restrict__ beff) {
  int h = blockIdx.x * 256 + threadIdx.x;
  if (h < HP) {
    float s = 0.f;
    if (h < H)
      for (int p = 0; p < P; ++p) s += Wb[p * H + h];
    weff[h] = s * (1.f / P);
    bap[h]  = (h < H) ? ba[h] : 0.f;
  }
  if (blockIdx.x == 0 && threadIdx.x == 0) {
    float s = 0.f;
    for (int p = 0; p < P; ++p) s += bb[p];
    *beff = s * (1.f / P);
  }
}

// ---------------------------------------------------------------------------
// Async stage of one m-tile's A fragments into LDS (double-buffered).
// ---------------------------------------------------------------------------
template <int KT>
__device__ __forceinline__ void stage_tile(const _Float16* __restrict__ apack,
                                           int mt, _Float16* sbuf, int tid) {
  const uint4* src = (const uint4*)(apack + (size_t)mt * KT * 512);
  uint4* dst = (uint4*)sbuf;
  for (int i = tid; i < KT * 64; i += 512) {
#if defined(HAVE_ASYNC_LDS)
    __builtin_amdgcn_global_load_async_to_lds_b128(
        (gvi4*)(src + i), (svi4*)(dst + i), 0, 0);
#else
    dst[i] = src[i];
#endif
  }
}

__device__ __forceinline__ void wait_async_copies() {
#if defined(HAVE_ASYNC_LDS)
#if defined(__has_builtin) && __has_builtin(__builtin_amdgcn_s_wait_asynccnt)
  __builtin_amdgcn_s_wait_asynccnt(0);
#else
  asm volatile("s_wait_asynccnt 0x0" ::: "memory");
#endif
#endif
}

// ---------------------------------------------------------------------------
// Branch GEMM: o[j] = wEff . relu(Wa m_j + ba) + bEff, 16 positions/wave,
// 16 waves/block sharing double-buffered async-staged A tiles in LDS.
// ---------------------------------------------------------------------------
template <int D, int KT, int MT>
__global__ __launch_bounds__(512) void branch_kernel(
    const _Float16* __restrict__ vh, const _Float16* __restrict__ avgh,
    const _Float16* __restrict__ apack, const float* __restrict__ weff,
    const float* __restrict__ bap, const float* __restrict__ beff,
    float* __restrict__ oraw) {
  constexpr int HP = MT * 16;
  __shared__ __align__(32) _Float16 sA[2][KT * 512];
  __shared__ float sW[HP];
  __shared__ float sB[HP];
  const int tid = threadIdx.x;
  for (int i = tid; i < HP; i += 512) {
    sW[i] = weff[i];
    sB[i] = bap[i];
  }

  const int lane = tid & 31;
  const int wave = tid >> 5;
  const int j0   = (blockIdx.x * 16 + wave) * 16;
  const int n    = lane & 15;
  const int klo  = (lane >> 4) << 4;  // K half held by this lane group
  const int j    = j0 + n;

  // Build B fragments: B[k][n] = feature (kt*32 + klo/2 + t + (t>=8 ? 8 : 0)) at position j.
  v16h bfrag[KT];
#pragma unroll
  for (int kt = 0; kt < KT; ++kt) {
#pragma unroll
    for (int t = 0; t < 16; ++t) {
      int f = kt * 32 + (klo >> 1) + t + (t >= 8 ? 8 : 0);
      _Float16 val = (_Float16)0.f;
      if (f < D - 1)        val = vh[(j - f) & L_MASK];   // circular roll
      else if (f == D - 1)  val = avgh[j];
      bfrag[kt][t] = val;
    }
  }

  // Prime the pipeline: stage tile 0 into buffer 0.
  stage_tile<KT>(apack, 0, sA[0], tid);

  float part = 0.f;
  const int rbase = (lane >> 4) << 3;  // C rows 0..7 or 8..15 per lane group
  for (int mt = 0; mt < MT; ++mt) {
    wait_async_copies();   // this wave's copies (incl. tile mt) complete
    __syncthreads();       // all waves' copies visible; prev compute done
    if (mt + 1 < MT)       // prefetch next tile into the other buffer
      stage_tile<KT>(apack, mt + 1, sA[(mt + 1) & 1], tid);

    const _Float16* buf = sA[mt & 1];
    v8f acc = {};
#pragma unroll
    for (int kt = 0; kt < KT; ++kt) {
      const v16h* ap = (const v16h*)(buf + kt * 512 + lane * 16);
      acc = __builtin_amdgcn_wmma_f32_16x16x32_f16(
          false, *ap, false, bfrag[kt], (short)0, acc, false, false);
    }
#pragma unroll
    for (int r = 0; r < 8; ++r) {
      int m = mt * 16 + rbase + r;
      float hv = acc[r] + sB[m];
      hv = hv > 0.f ? hv : 0.f;
      part += hv * sW[m];
    }
  }
  part += __shfl_xor(part, 16, 32);  // fold row halves (same column n)
  if (lane < 16) oraw[j] = part + beff[0];
}

// ---------------------------------------------------------------------------
// t1 = o1raw / energy ; final LN write
// ---------------------------------------------------------------------------
__global__ __launch_bounds__(256) void divide_kernel(
    const float* __restrict__ o1raw, const float* __restrict__ energy,
    float* __restrict__ t1) {
  int i = blockIdx.x * 256 + threadIdx.x;
  t1[i] = o1raw[i] / energy[i];
}

__global__ __launch_bounds__(256) void ln_write_kernel(
    const float* __restrict__ src, const float* __restrict__ stats,
    float* __restrict__ dst) {
  int i = blockIdx.x * 256 + threadIdx.x;
  dst[i] = (src[i] - stats[0]) * stats[1];
}

// ---------------------------------------------------------------------------
extern "C" void kernel_launch(void* const* d_in, const int* in_sizes, int n_in,
                              void* d_out, int out_size, void* d_ws,
                              size_t ws_size, hipStream_t stream) {
  (void)in_sizes; (void)n_in; (void)out_size; (void)ws_size;
  constexpr size_t L = L_TOTAL;
  // Branch shapes
  constexpr int D1 = 65,  H1 = 260, KT1 = 3, MT1 = 17, HP1 = 272, P1 = 3;
  constexpr int D2 = 129, H2 = 516, KT2 = 5, MT2 = 33, HP2 = 528, P2 = 21;

  const float* x   = (const float*)d_in[0];
  const float* W1a = (const float*)d_in[1];
  const float* b1a = (const float*)d_in[2];
  const float* W1b = (const float*)d_in[3];
  const float* b1b = (const float*)d_in[4];
  const float* W2a = (const float*)d_in[5];
  const float* b2a = (const float*)d_in[6];
  const float* W2b = (const float*)d_in[7];
  const float* b2b = (const float*)d_in[8];
  float* out = (float*)d_out;

  // Workspace layout (all 16B aligned)
  char* ws = (char*)d_ws;
  constexpr size_t OFF_VF   = 0;                           // L f32
  constexpr size_t OFF_AVGF = OFF_VF   + L * 4;            // L f32
  constexpr size_t OFF_EN   = OFF_AVGF + L * 4;            // L f32
  constexpr size_t OFF_O1   = OFF_EN   + L * 4;            // L f32
  constexpr size_t OFF_O2   = OFF_O1   + L * 4;            // L f32
  constexpr size_t OFF_T1   = OFF_O2   + L * 4;            // L f32
  constexpr size_t OFF_VH   = OFF_T1   + L * 4;            // L f16
  constexpr size_t OFF_AVGH = OFF_VH   + L * 2;            // L f16
  constexpr size_t OFF_A1   = OFF_AVGH + L * 2;            // MT1*KT1*512 f16
  constexpr size_t OFF_A2   = OFF_A1 + (size_t)MT1 * KT1 * 512 * 2;
  constexpr size_t OFF_W1   = OFF_A2 + (size_t)MT2 * KT2 * 512 * 2;
  constexpr size_t OFF_BA1  = OFF_W1  + HP1 * 4;
  constexpr size_t OFF_W2   = OFF_BA1 + HP1 * 4;
  constexpr size_t OFF_BA2  = OFF_W2  + HP2 * 4;
  constexpr size_t OFF_BE1  = OFF_BA2 + HP2 * 4;
  constexpr size_t OFF_BE2  = OFF_BE1 + 256;
  constexpr size_t OFF_STX  = OFF_BE2 + 256;               // mean/rstd x
  constexpr size_t OFF_ST1  = OFF_STX + 256;               // mean/rstd t1
  constexpr size_t OFF_ST2  = OFF_ST1 + 256;               // mean/rstd t2
  constexpr size_t OFF_PX   = OFF_ST2 + 256;               // 256*2 f32
  constexpr size_t OFF_P1   = OFF_PX  + 2048;
  constexpr size_t OFF_P2   = OFF_P1  + 2048;

  float*    vf    = (float*)(ws + OFF_VF);
  float*    avgf  = (float*)(ws + OFF_AVGF);
  float*    energy= (float*)(ws + OFF_EN);
  float*    o1raw = (float*)(ws + OFF_O1);
  float*    o2raw = (float*)(ws + OFF_O2);
  float*    t1    = (float*)(ws + OFF_T1);
  _Float16* vh    = (_Float16*)(ws + OFF_VH);
  _Float16* avgh  = (_Float16*)(ws + OFF_AVGH);
  _Float16* a1    = (_Float16*)(ws + OFF_A1);
  _Float16* a2    = (_Float16*)(ws + OFF_A2);
  float*    w1    = (float*)(ws + OFF_W1);
  float*    ba1p  = (float*)(ws + OFF_BA1);
  float*    w2    = (float*)(ws + OFF_W2);
  float*    ba2p  = (float*)(ws + OFF_BA2);
  float*    be1   = (float*)(ws + OFF_BE1);
  float*    be2   = (float*)(ws + OFF_BE2);
  float*    stX   = (float*)(ws + OFF_STX);
  float*    st1   = (float*)(ws + OFF_ST1);
  float*    st2   = (float*)(ws + OFF_ST2);
  float*    pX    = (float*)(ws + OFF_PX);
  float*    p1    = (float*)(ws + OFF_P1);
  float*    p2    = (float*)(ws + OFF_P2);

  const dim3 blk(256);
  const float invL = 1.f / (float)L;

  // 1) LayerNorm stats of x, normalize
  stat_partial_kernel<<<256, blk, 0, stream>>>(x, (int)L, pX);
  stat_final_kernel<<<1, blk, 0, stream>>>(pX, 256, invL, stX);
  normalize_kernel<<<L / 256, blk, 0, stream>>>(x, stX, vf, vh);

  // 2) avg / energy pooling
  avgpool_kernel<<<L / 256, blk, 0, stream>>>(vf, avgf, avgh, energy);

  // 3) weight prep
  pack_a_kernel<D1, H1, KT1, MT1>
      <<<(MT1 * KT1 * 512 + 255) / 256, blk, 0, stream>>>(W1a, a1);
  pack_a_kernel<D2, H2, KT2, MT2>
      <<<(MT2 * KT2 * 512 + 255) / 256, blk, 0, stream>>>(W2a, a2);
  weff_kernel<H1, HP1, P1><<<(HP1 + 255) / 256, blk, 0, stream>>>(
      W1b, b1b, b1a, w1, ba1p, be1);
  weff_kernel<H2, HP2, P2><<<(HP2 + 255) / 256, blk, 0, stream>>>(
      W2b, b2b, b2a, w2, ba2p, be2);

  // 4) WMMA branch GEMMs (16 waves/block * 16 positions/wave = 256 pos/block)
  branch_kernel<D1, KT1, MT1><<<L / 256, dim3(512), 0, stream>>>(
      vh, avgh, a1, w1, ba1p, be1, o1raw);
  branch_kernel<D2, KT2, MT2><<<L / 256, dim3(512), 0, stream>>>(
      vh, avgh, a2, w2, ba2p, be2, o2raw);

  // 5) o1/energy, final LayerNorms, write outputs
  divide_kernel<<<L / 256, blk, 0, stream>>>(o1raw, energy, t1);
  stat_partial_kernel<<<256, blk, 0, stream>>>(t1, (int)L, p1);
  stat_final_kernel<<<1, blk, 0, stream>>>(p1, 256, invL, st1);
  stat_partial_kernel<<<256, blk, 0, stream>>>(o2raw, (int)L, p2);
  stat_final_kernel<<<1, blk, 0, stream>>>(p2, 256, invL, st2);
  ln_write_kernel<<<L / 256, blk, 0, stream>>>(t1, st1, out);
  ln_write_kernel<<<L / 256, blk, 0, stream>>>(o2raw, st2, out + L);
}
